// FlaxRwkvSelfAttention_76879914598667
// MI455X (gfx1250) — matrix operand order repaired and checked by reading, not compiled, not run on hardware
//
#include <hip/hip_runtime.h>
#include <hip/hip_bf16.h>
#include <math.h>

#define TSEQ   4096
#define HDIM   2048
#define CHUNKS 64
#define CLEN   (TSEQ / CHUNKS)    // 64

typedef __bf16 bf16_t;
typedef bf16_t v16bf __attribute__((ext_vector_type(16)));
typedef float  v8f   __attribute__((ext_vector_type(8)));
typedef unsigned int u32x4 __attribute__((ext_vector_type(4)));
typedef int          i32x4 __attribute__((ext_vector_type(4)));
typedef int          i32x8 __attribute__((ext_vector_type(8)));

#if defined(__has_builtin)
#if __has_builtin(__builtin_amdgcn_tensor_load_to_lds) && \
    __has_builtin(__builtin_amdgcn_s_wait_tensorcnt)
#define USE_TDM 1
#endif
#endif

// ---------- helpers ----------

__device__ __forceinline__ unsigned short f2bf(float x) {
    // round-to-nearest-even float32 -> bf16
    unsigned int u = __float_as_uint(x);
    unsigned int r = (u + 0x7FFFu + ((u >> 16) & 1u)) >> 16;
    return (unsigned short)r;
}

union FragU {
    v16bf v;
    uint4 q[2];
};

// A-fragment (16x32 bf16, MxK): lane holds row m = lane&15, half = lane>>4.
// VGPR0..3 = K [half*8 .. +7], VGPR4..7 = K [16+half*8 .. +7]
__device__ __forceinline__ v16bf load_frag_a(const unsigned short* p, int half) {
    FragU f;
    f.q[0] = *(const uint4*)(p + half * 8);
    f.q[1] = *(const uint4*)(p + 16 + half * 8);
    return f.v;
}

// B-fragment (32x16 bf16, KxN) read from W^T rows: lane holds col n = lane&15.
// VGPR0..3 = K [half*8 .. +7], VGPR4..7 = K [16+half*8 .. +7]
__device__ __forceinline__ v16bf load_frag_b(const unsigned short* p, int half) {
    FragU f;
    f.q[0] = *(const uint4*)(p + half * 8);
    f.q[1] = *(const uint4*)(p + 16 + half * 8);
    return f.v;
}

#ifdef USE_TDM
// Issue one 2D TDM tile load: global (row-major, rowStride elems of 2B) -> LDS
// (tile rows contiguous). D# bitfields per CDNA5 ISA.
__device__ __forceinline__ void tdm_load_2d(unsigned lds_off,
                                            const unsigned short* gptr,
                                            int tile_w, int tile_h,
                                            int tensor_h, int row_stride) {
    unsigned long long ga = (unsigned long long)(size_t)gptr;
    u32x4 g0;
    g0[0] = 1u;                                   // count=1 (valid), user mode
    g0[1] = lds_off;                              // lds_addr
    g0[2] = (unsigned)(ga & 0xFFFFFFFFu);         // global_addr[31:0]
    g0[3] = (unsigned)((ga >> 32) & 0x01FFFFFFu)  // global_addr[56:32]
          | (2u << 30);                           // type = 2 ("image")
    i32x8 g1;
    g1[0] = (1 << 16);                            // wg_mask=0, data_size=1 (2B)
    g1[1] = (int)((unsigned)(row_stride & 0xFFFF) << 16);          // tensor_dim0 lo
    g1[2] = (int)(((unsigned)row_stride >> 16) |
                  ((unsigned)(tensor_h & 0xFFFF) << 16));          // dim0 hi | dim1 lo
    g1[3] = (int)(((unsigned)tensor_h >> 16) |
                  ((unsigned)tile_w << 16));                       // dim1 hi | tile_dim0
    g1[4] = tile_h;                               // tile_dim1, tile_dim2=0
    g1[5] = row_stride;                           // tensor_dim0_stride[31:0]
    g1[6] = 0;                                    // stride hi, dim1_stride lo
    g1[7] = 0;
    i32x4 z4 = { 0, 0, 0, 0 };
#if __clang_major__ >= 23
    i32x8 z8 = { 0, 0, 0, 0, 0, 0, 0, 0 };
    __builtin_amdgcn_tensor_load_to_lds(g0, g1, z4, z4, z8, 0);
#else
    __builtin_amdgcn_tensor_load_to_lds(g0, g1, z4, z4, 0);
#endif
}
#endif

// ---------- kernel 1: token shift + time-mix, fp32 -> bf16 ----------

__global__ __launch_bounds__(256) void rwkv_mix_kernel(
    const float* __restrict__ hidden, const float* __restrict__ sx,
    const float* __restrict__ tmk, const float* __restrict__ tmv,
    const float* __restrict__ tmr,
    unsigned short* __restrict__ kx, unsigned short* __restrict__ vx,
    unsigned short* __restrict__ rx)
{
    int idx = blockIdx.x * blockDim.x + threadIdx.x;   // over T*H
    int t = idx >> 11;            // H == 2048
    int h = idx & (HDIM - 1);
    float cur  = hidden[idx];
    float prev = (t == 0) ? sx[h] : hidden[idx - HDIM];
    float mk = tmk[h], mv = tmv[h], mr = tmr[h];
    kx[idx] = f2bf(cur * mk + prev * (1.0f - mk));
    vx[idx] = f2bf(cur * mv + prev * (1.0f - mv));
    rx[idx] = f2bf(cur * mr + prev * (1.0f - mr));
}

// ---------- kernel 2: W [K,N] fp32 -> W^T [N,K] bf16 (LDS tiled transpose) ----------

__global__ __launch_bounds__(256) void rwkv_packw_kernel(
    const float* __restrict__ W, unsigned short* __restrict__ Wt)
{
    __shared__ unsigned short tile[32][33];
    int bk = blockIdx.y * 32;
    int bn = blockIdx.x * 32;
    int tx = threadIdx.x & 31;
    int ty = threadIdx.x >> 5;
    #pragma unroll
    for (int r = ty; r < 32; r += 8)
        tile[r][tx] = f2bf(W[(size_t)(bk + r) * HDIM + bn + tx]);
    __syncthreads();
    #pragma unroll
    for (int r = ty; r < 32; r += 8)
        Wt[(size_t)(bn + r) * HDIM + bk + tx] = tile[tx][r];
}

// ---------- kernel 3: WMMA GEMM via TDM double-buffered LDS staging ----------
// C[T,H] = A[T,H](bf16) x W^T[N,K](bf16).  Block tile 128(M) x 64(N), K-step 64.
// mode 0: C = A*W;  mode 1: C = sigmoid(A*W);  mode 2: C = addSrc + A*W

__global__ __launch_bounds__(256) void rwkv_gemm_kernel(
    const unsigned short* __restrict__ Ab,
    const unsigned short* __restrict__ Wt,
    const float* __restrict__ addSrc,
    float* __restrict__ C,
    int mode)
{
    __shared__ __align__(16) unsigned short ldsA[2][128 * 64];   // 32 KB
    __shared__ __align__(16) unsigned short ldsB[2][64 * 64];    // 16 KB

    const int lane = threadIdx.x & 31;
    const int wave = threadIdx.x >> 5;        // 0..7
    const int wm = wave >> 1;                 // 4 waves in M
    const int wn = wave & 1;                  // 2 waves in N
    const int gM = blockIdx.y * 128;
    const int gN = blockIdx.x * 64;
    const int half = lane >> 4;
    const int lm = lane & 15;

    v8f acc00 = {}, acc01 = {}, acc10 = {}, acc11 = {};

    const int NKI = HDIM / 64;                // 32 K-iterations

#ifdef USE_TDM
    const bool issuer = (threadIdx.x < 32);   // wave 0 drives the TDM
    if (issuer) {
        tdm_load_2d((unsigned)(size_t)&ldsA[0][0], Ab + (size_t)gM * HDIM,
                    64, 128, TSEQ, HDIM);
        tdm_load_2d((unsigned)(size_t)&ldsB[0][0], Wt + (size_t)gN * HDIM,
                    64, 64, HDIM, HDIM);
    }
#endif

    for (int i = 0; i < NKI; ++i) {
        const int cur = i & 1;
#ifdef USE_TDM
        if (i + 1 < NKI) {
            if (issuer) {
                int k0n = (i + 1) * 64;
                tdm_load_2d((unsigned)(size_t)&ldsA[cur ^ 1][0],
                            Ab + (size_t)gM * HDIM + k0n, 64, 128, TSEQ, HDIM);
                tdm_load_2d((unsigned)(size_t)&ldsB[cur ^ 1][0],
                            Wt + (size_t)gN * HDIM + k0n, 64, 64, HDIM, HDIM);
                __builtin_amdgcn_s_wait_tensorcnt(2);  // current buffer done
            }
        } else {
            if (issuer) __builtin_amdgcn_s_wait_tensorcnt(0);
        }
        __syncthreads();
#else
        // synchronous cooperative staging fallback
        {
            const int k0 = i * 64;
            for (int e = threadIdx.x; e < 128 * 64 / 8; e += 256) {
                int row = e >> 3, seg = e & 7;
                *(uint4*)&ldsA[cur][row * 64 + seg * 8] =
                    *(const uint4*)(Ab + (size_t)(gM + row) * HDIM + k0 + seg * 8);
            }
            for (int e = threadIdx.x; e < 64 * 64 / 8; e += 256) {
                int row = e >> 3, seg = e & 7;
                *(uint4*)&ldsB[cur][row * 64 + seg * 8] =
                    *(const uint4*)(Wt + (size_t)(gN + row) * HDIM + k0 + seg * 8);
            }
        }
        __syncthreads();
#endif
        const unsigned short* aR0 = &ldsA[cur][(wm * 32 + lm) * 64];
        const unsigned short* aR1 = aR0 + 16 * 64;
        const unsigned short* bR0 = &ldsB[cur][(wn * 32 + lm) * 64];
        const unsigned short* bR1 = bR0 + 16 * 64;
        #pragma unroll
        for (int kk = 0; kk < 64; kk += 32) {
            v16bf a0 = load_frag_a(aR0 + kk, half);
            v16bf a1 = load_frag_a(aR1 + kk, half);
            v16bf b0 = load_frag_b(bR0 + kk, half);
            v16bf b1 = load_frag_b(bR1 + kk, half);
            acc00 = __builtin_amdgcn_wmma_f32_16x16x32_bf16(false, a0, false, b0,
                                                            (short)0, acc00, false, false);
            acc01 = __builtin_amdgcn_wmma_f32_16x16x32_bf16(false, a0, false, b1,
                                                            (short)0, acc01, false, false);
            acc10 = __builtin_amdgcn_wmma_f32_16x16x32_bf16(false, a1, false, b0,
                                                            (short)0, acc10, false, false);
            acc11 = __builtin_amdgcn_wmma_f32_16x16x32_bf16(false, a1, false, b1,
                                                            (short)0, acc11, false, false);
        }
        __syncthreads();   // everyone done reading before next overwrite
    }

    // C/D layout: VGPR r of lane l -> row (half*8 + r), col (l&15)
    v8f accs[2][2] = { { acc00, acc01 }, { acc10, acc11 } };
    #pragma unroll
    for (int im = 0; im < 2; ++im) {
        #pragma unroll
        for (int in = 0; in < 2; ++in) {
            int col = gN + wn * 32 + in * 16 + lm;
            #pragma unroll
            for (int r = 0; r < 8; ++r) {
                int row = gM + wm * 32 + im * 16 + half * 8 + r;
                size_t idx = (size_t)row * HDIM + col;
                float val = accs[im][in][r];
                if (mode == 1) {
                    val = 1.0f / (1.0f + __expf(-val));
                } else if (mode == 2) {
                    val = addSrc[idx] + val;
                }
                C[idx] = val;
            }
        }
    }
}

// ---------- chunked parallel WKV scan ----------
// Step: S' = decay(S) + u, stabilized as (aa,bb,pp).  A length-L chunk acts as
// "pp += L*w, then stabilized-add the chunk's zero-seeded accumulator", which is
// the same formula as a single step, so the composition is exact.

// phase 1: per (chunk, channel) accumulate the inhomogeneous term from identity
__global__ __launch_bounds__(256) void rwkv_scan1_kernel(
    const float* __restrict__ kbuf, const float* __restrict__ vbuf,
    const float* __restrict__ time_decay,
    float* __restrict__ accA, float* __restrict__ accB, float* __restrict__ accP)
{
    int c = blockIdx.x >> 3;                          // 8 blocks per chunk
    int h = ((blockIdx.x & 7) << 8) + threadIdx.x;
    const float wd = -__expf(time_decay[h]);

    float aa = 0.0f, bb = 0.0f, pp = -1.0e30f;        // identity accumulator
    int t0 = c * CLEN;
    for (int t = t0; t < t0 + CLEN; ++t) {
        size_t i = (size_t)t * HDIM + h;
        float kk = kbuf[i], vv = vbuf[i];
        float ww2 = wd + pp;
        float p2  = fmaxf(ww2, kk);
        float e1b = __expf(ww2 - p2);
        float e2b = __expf(kk - p2);
        aa = e1b * aa + e2b * vv;
        bb = e1b * bb + e2b;
        pp = p2;
    }
    size_t o = (size_t)c * HDIM + h;
    accA[o] = aa; accB[o] = bb; accP[o] = pp;
}

// phase 2: exclusive scan of chunk operators per channel; emits per-chunk
// incoming states and the final (aa,bb,pp) + hidden[-1] into d_out tail.
__global__ __launch_bounds__(256) void rwkv_scan2_kernel(
    const float* __restrict__ accA, const float* __restrict__ accB,
    const float* __restrict__ accP,
    float* __restrict__ stA, float* __restrict__ stB, float* __restrict__ stP,
    const float* __restrict__ time_decay,
    const float* __restrict__ aa_in, const float* __restrict__ bb_in,
    const float* __restrict__ pp_in,
    const float* __restrict__ hidden,
    float* __restrict__ outTail)   // [x_last | aa | bb | pp]
{
    int h = blockIdx.x * blockDim.x + threadIdx.x;
    if (h >= HDIM) return;
    const float wd = -__expf(time_decay[h]);
    const float wL = (float)CLEN * wd;

    float aa = aa_in[h], bb = bb_in[h], pp = pp_in[h];
    for (int c = 0; c < CHUNKS; ++c) {
        size_t o = (size_t)c * HDIM + h;
        stA[o] = aa; stB[o] = bb; stP[o] = pp;        // incoming state of chunk c
        float ww2 = wL + pp;                          // decay by L steps
        float pc  = accP[o];
        float p2  = fmaxf(ww2, pc);
        float e1  = __expf(ww2 - p2);
        float e2  = __expf(pc - p2);
        aa = e1 * aa + e2 * accA[o];
        bb = e1 * bb + e2 * accB[o];
        pp = p2;
    }
    outTail[h]            = hidden[(size_t)(TSEQ - 1) * HDIM + h];
    outTail[HDIM + h]     = aa;
    outTail[2 * HDIM + h] = bb;
    outTail[3 * HDIM + h] = pp;
}

// phase 3: seeded replay of each chunk, emitting bf16(r * wkv)
__global__ __launch_bounds__(256) void rwkv_scan3_kernel(
    const float* __restrict__ kbuf, const float* __restrict__ vbuf,
    const float* __restrict__ rbuf,
    const float* __restrict__ time_decay, const float* __restrict__ time_first,
    const float* __restrict__ stA, const float* __restrict__ stB,
    const float* __restrict__ stP,
    unsigned short* __restrict__ rwkv_bf)
{
    int c = blockIdx.x >> 3;
    int h = ((blockIdx.x & 7) << 8) + threadIdx.x;
    const float tf = time_first[h];
    const float wd = -__expf(time_decay[h]);

    size_t o = (size_t)c * HDIM + h;
    float aa = stA[o], bb = stB[o], pp = stP[o];

    int t0 = c * CLEN;
    for (int t = t0; t < t0 + CLEN; ++t) {
        size_t i = (size_t)t * HDIM + h;
        float kk = kbuf[i], vv = vbuf[i];

        float ww = tf + kk;
        float p  = fmaxf(pp, ww);
        float e1 = __expf(pp - p);
        float e2 = __expf(ww - p);
        float out_t = (e1 * aa + e2 * vv) / (e1 * bb + e2);

        float ww2 = wd + pp;
        float p2  = fmaxf(ww2, kk);
        float e1b = __expf(ww2 - p2);
        float e2b = __expf(kk - p2);
        aa = e1b * aa + e2b * vv;
        bb = e1b * bb + e2b;
        pp = p2;

        rwkv_bf[i] = f2bf(rbuf[i] * out_t);
    }
}

// ---------- host launcher ----------

extern "C" void kernel_launch(void* const* d_in, const int* in_sizes, int n_in,
                              void* d_out, int out_size, void* d_ws, size_t ws_size,
                              hipStream_t stream) {
    const float* hidden     = (const float*)d_in[0];
    const float* sx         = (const float*)d_in[1];
    const float* aa         = (const float*)d_in[2];
    const float* bb         = (const float*)d_in[3];
    const float* pp         = (const float*)d_in[4];
    const float* time_decay = (const float*)d_in[5];
    const float* time_first = (const float*)d_in[6];
    const float* tmk        = (const float*)d_in[7];
    const float* tmv        = (const float*)d_in[8];
    const float* tmr        = (const float*)d_in[9];
    const float* Wk         = (const float*)d_in[10];
    const float* Wv         = (const float*)d_in[11];
    const float* Wr         = (const float*)d_in[12];
    const float* Wo         = (const float*)d_in[13];
    float* out = (float*)d_out;

    const size_t TH = (size_t)TSEQ * HDIM;
    const size_t HH = (size_t)HDIM * HDIM;
    const size_t CH = (size_t)CHUNKS * HDIM;

    char* p = (char*)d_ws;
    unsigned short* kx  = (unsigned short*)p; p += TH * 2;
    unsigned short* vx  = (unsigned short*)p; p += TH * 2;
    unsigned short* rx  = (unsigned short*)p; p += TH * 2;
    unsigned short* WkT = (unsigned short*)p; p += HH * 2;
    unsigned short* WvT = (unsigned short*)p; p += HH * 2;
    unsigned short* WrT = (unsigned short*)p; p += HH * 2;
    unsigned short* WoT = (unsigned short*)p; p += HH * 2;
    float* kbuf = (float*)p; p += TH * 4;
    float* vbuf = (float*)p; p += TH * 4;
    float* rbuf = (float*)p; p += TH * 4;
    float* accA = (float*)p; p += CH * 4;
    float* accB = (float*)p; p += CH * 4;
    float* accP = (float*)p; p += CH * 4;
    float* stA  = (float*)p; p += CH * 4;
    float* stB  = (float*)p; p += CH * 4;
    float* stP  = (float*)p; p += CH * 4;
    unsigned short* rwkv = kx;   // alias: key_x dead after the k-GEMM

    // 1) token shift + mix -> bf16
    rwkv_mix_kernel<<<dim3((unsigned)(TH / 256)), 256, 0, stream>>>(
        hidden, sx, tmk, tmv, tmr, kx, vx, rx);

    // 2) pack weights: fp32 [K,N] -> bf16 [N,K]
    dim3 pgrid(HDIM / 32, HDIM / 32);
    rwkv_packw_kernel<<<pgrid, 256, 0, stream>>>(Wk, WkT);
    rwkv_packw_kernel<<<pgrid, 256, 0, stream>>>(Wv, WvT);
    rwkv_packw_kernel<<<pgrid, 256, 0, stream>>>(Wr, WrT);
    rwkv_packw_kernel<<<pgrid, 256, 0, stream>>>(Wo, WoT);

    // 3) WMMA GEMMs: k, v, r = sigmoid(.)
    dim3 ggrid(HDIM / 64, TSEQ / 128);
    rwkv_gemm_kernel<<<ggrid, 256, 0, stream>>>(kx, WkT, nullptr, kbuf, 0);
    rwkv_gemm_kernel<<<ggrid, 256, 0, stream>>>(vx, WvT, nullptr, vbuf, 0);
    rwkv_gemm_kernel<<<ggrid, 256, 0, stream>>>(rx, WrT, nullptr, rbuf, 1);

    // 4) chunked parallel WKV scan
    rwkv_scan1_kernel<<<dim3(CHUNKS * (HDIM / 256)), 256, 0, stream>>>(
        kbuf, vbuf, time_decay, accA, accB, accP);
    rwkv_scan2_kernel<<<dim3(HDIM / 256), 256, 0, stream>>>(
        accA, accB, accP, stA, stB, stP, time_decay, aa, bb, pp,
        hidden, out + TH);
    rwkv_scan3_kernel<<<dim3(CHUNKS * (HDIM / 256)), 256, 0, stream>>>(
        kbuf, vbuf, rbuf, time_decay, time_first, stA, stB, stP, rwkv);

    // 5) out = hidden + (r*wkv) @ Wo
    rwkv_gemm_kernel<<<ggrid, 256, 0, stream>>>(rwkv, WoT, hidden, out, 2);
}
